// DecoderLSTMCell_87969520156821
// MI455X (gfx1250) — compile-verified, matrix operand order checked by reading, not run on hardware
//
#include <hip/hip_runtime.h>

// Problem dims (fixed by the reference).
#define B_DIM   16384
#define IN_DIM  1024
#define H_DIM   1024
#define CTX_DIM 2048
#define K_TOT   (IN_DIM + H_DIM + CTX_DIM)   // 4096

// Tiling: block = 256 threads = 8 wave32.
// A block computes a [BM x BN] tile for ALL FOUR gates so the LSTM combine
// is fused in-register (the 4 gates of one (m,n) live in the same lane slot).
#define BM 128
#define BN 32            // per gate

// Fast path: BK2 = 64 (two WMMA K-slices per staged buffer / barrier).
#define BK2 64
#define LDS_STRIDE2 80   // bf16 elems per LDS row (64 data + 16 pad), 160B rows
#define TILE2_U (BM * LDS_STRIDE2 / 2)   // 5120 uints = 20KB per buffer

// Fallback path: BK = 32.
#define BK 32
#define LDS_STRIDE 48    // bf16 elems per LDS row (32 data + 16 pad), 96B rows
#define TILE_U (BM * LDS_STRIDE / 2)     // 3072 uints = 12KB
#define BUF_U  (2 * TILE_U)              // A-tile + B-tile per buffer

typedef __attribute__((ext_vector_type(16))) __bf16 bf16x16;
typedef __attribute__((ext_vector_type(8)))  float  f32x8;

#define WMMA_BF16(A, Bf, Cacc) \
  __builtin_amdgcn_wmma_f32_16x16x32_bf16(false, (A), false, (Bf), (short)0, (Cacc), false, false)

// ---- cheap f32 -> packed bf16x2: two v_add (round-to-nearest) + one v_perm_b32
static __device__ __forceinline__ unsigned pack2bf(float lo, float hi) {
  unsigned ul = __builtin_bit_cast(unsigned, lo) + 0x8000u;
  unsigned uh = __builtin_bit_cast(unsigned, hi) + 0x8000u;
  return __builtin_amdgcn_perm(uh, ul, 0x07060302u);   // { uh[31:16], ul[31:16] }
}

static __device__ __forceinline__ float fast_exp(float x) {
  return __builtin_amdgcn_exp2f(x * 1.44269504088896341f);   // v_exp_f32
}
static __device__ __forceinline__ float fast_sig(float x) {
  return __builtin_amdgcn_rcpf(1.0f + fast_exp(-x));
}
static __device__ __forceinline__ float fast_tanh(float x) {
  return 1.0f - 2.0f * __builtin_amdgcn_rcpf(fast_exp(2.0f * x) + 1.0f);
}

// Async global->LDS 16B copy (CDNA5 path, tracked by ASYNCcnt).
static __device__ __forceinline__ void async_cp16(unsigned lds_off, const void* gsrc) {
  asm volatile("global_load_async_to_lds_b128 %0, %1, off"
               :: "v"(lds_off), "v"(gsrc) : "memory");
}
static __device__ __forceinline__ void wait_async0() {
  asm volatile("s_wait_asynccnt 0x0" ::: "memory");
}

// One WMMA K-slice: 8 B fragments from LDS (row stride SU uints),
// interleaved with in-place WMMAs using two rotating temps.
template <int SU>
static __device__ __forceinline__ void wmma_step(const bf16x16 af,
    const unsigned* __restrict__ lb, f32x8 acc[8])
{
#define LDB(t) (*(const bf16x16*)(lb + (t) * 16 * SU))
  bf16x16 t0 = LDB(0);
  bf16x16 t1 = LDB(1);
  acc[0] = WMMA_BF16(af, t0, acc[0]);  t0 = LDB(2);
  acc[1] = WMMA_BF16(af, t1, acc[1]);  t1 = LDB(3);
  acc[2] = WMMA_BF16(af, t0, acc[2]);  t0 = LDB(4);
  acc[3] = WMMA_BF16(af, t1, acc[3]);  t1 = LDB(5);
  acc[4] = WMMA_BF16(af, t0, acc[4]);  t0 = LDB(6);
  acc[5] = WMMA_BF16(af, t1, acc[5]);  t1 = LDB(7);
  acc[6] = WMMA_BF16(af, t0, acc[6]);
  acc[7] = WMMA_BF16(af, t1, acc[7]);
#undef LDB
}

// ------------------------------------------------------------------
// Shared LSTM epilogue. acc[t], t = 2*gate + ntile.
// C layout: VGPR v, lane l -> row = v + 8*(l>=16), col = l&15.
// ------------------------------------------------------------------
static __device__ __forceinline__ void lstm_epilogue(
    f32x8 acc[8], const float* __restrict__ c0,
    const float* __restrict__ bias,
    float* __restrict__ out_c, float* __restrict__ out_h,
    int m0, int n0, int wv, int l16, int half)
{
#pragma unroll
  for (int nt = 0; nt < 2; ++nt) {
    const int n   = n0 + nt * 16 + l16;
    const float bi  = bias[0 * H_DIM + n];
    const float bff = bias[1 * H_DIM + n];
    const float bo  = bias[2 * H_DIM + n];
    const float bc  = bias[3 * H_DIM + n];
#pragma unroll
    for (int v = 0; v < 8; ++v) {
      const int m = m0 + wv * 16 + v + 8 * half;
      const size_t idx = (size_t)m * H_DIM + n;
      const float gi = acc[0 + nt][v] + bi;
      const float gf = acc[2 + nt][v] + bff;
      const float go = acc[4 + nt][v] + bo;
      const float gc = acc[6 + nt][v] + bc;
      const float iv = fast_sig(gi);
      const float fv = fast_sig(gf);
      const float ov = fast_sig(go);
      const float ch = fast_tanh(gc);
      const float cv = iv * ch + fv * c0[idx];
      const float hv = ov * fast_tanh(cv);
      out_c[idx] = cv;
      out_h[idx] = hv;
    }
  }
}

// ==================================================================
// Pre-pass: convert [rows x 4096] K-concatenated f32 sources
// (strides 1024 | 1024 | 2048) into packed bf16, row-major stride 4096.
// ==================================================================
__global__ __launch_bounds__(256)
void convert_cat_bf16(const float* __restrict__ s0, const float* __restrict__ s1,
                      const float* __restrict__ s2, unsigned* __restrict__ dst)
{
  const size_t t = (size_t)blockIdx.x * 256 + threadIdx.x;
  const size_t e = t * 4;                    // element index in [rows][4096]
  const int    k = (int)(e & 4095u);
  const size_t r = e >> 12;
  const float* src; int kk;
  if (k < 1024)      { src = s0 + r * 1024; kk = k; }
  else if (k < 2048) { src = s1 + r * 1024; kk = k - 1024; }
  else               { src = s2 + r * 2048; kk = k - 2048; }
  const float4 v = *(const float4*)(src + kk);
  uint2 o;
  o.x = pack2bf(v.x, v.y);
  o.y = pack2bf(v.z, v.w);
  *(uint2*)(dst + (e >> 1)) = o;
}

// ==================================================================
// Fast path: A and B already bf16 in workspace.
//  - BK2=64: 16 WMMAs per barrier (halved sync/loop overhead).
//  - A fragments directly from global (disjoint per-wave stripes; bf16
//    row-major == A-fragment layout); prefetched one stage ahead.
//  - B tile shared by all 8 waves -> async global->LDS staging (double
//    buffered, 64B/thread/stage). Zero conversion VALU in the loop.
//  - K loop NOT unrolled, last stage peeled: branch-free body, one
//    stable register home per accumulator (in-place C->D WMMA).
// ==================================================================
__global__ __launch_bounds__(256)
void lstm_wmma_pre(const unsigned* __restrict__ Abf,   // [B_DIM][4096] bf16
                   const unsigned* __restrict__ Bbf,   // [4096][4096] bf16
                   const float* __restrict__ c0, const float* __restrict__ bias,
                   float* __restrict__ out_c, float* __restrict__ out_h)
{
  __shared__ __attribute__((aligned(32))) unsigned lds[2 * TILE2_U];   // 40 KB

  const int tid  = threadIdx.x;
  const int lane = tid & 31;
  const int wv   = tid >> 5;
  const int l16  = lane & 15;
  const int half = lane >> 4;
  const int m0   = blockIdx.y * BM;
  const int n0   = blockIdx.x * BN;

  f32x8 acc[8];
#pragma unroll
  for (int t = 0; t < 8; ++t)
    acc[t] = (f32x8){0.f,0.f,0.f,0.f,0.f,0.f,0.f,0.f};

  // B staging: 256 threads x 64B cover the 128x64 bf16 tile per stage.
  const int brow = tid >> 1;                 // LDS row 0..127 = gate*32 + rr
  const int bk16 = (tid & 1) * 32;           // bf16 offset within BK2
  const unsigned growB = (unsigned)((brow >> 5) * H_DIM + n0 + (brow & 31));
  const char* bsrc = (const char*)Bbf + (size_t)growB * (K_TOT * 2) + bk16 * 2;

  // Wave-relative LDS byte address of this thread's staging slot.
  const unsigned lds0  = (unsigned)(size_t)(void*)&lds[0];
  const unsigned bdstb = lds0 + (unsigned)((brow * LDS_STRIDE2 + bk16) * 2);

  // A fragment source (direct global).
  const unsigned arow = (unsigned)(m0 + wv * 16 + l16);
  const unsigned* asrc = Abf + (((size_t)arow * K_TOT + half * 16) >> 1);

  // Per-lane base of the B fragments in LDS (uint units).
  const int bf_lane = l16 * (LDS_STRIDE2 / 2) + half * 8;

  // Prologue: async-stage B for stage 0 into buffer 0 (4 x 16B).
  async_cp16(bdstb,      bsrc);
  async_cp16(bdstb + 16, bsrc + 16);
  async_cp16(bdstb + 32, bsrc + 32);
  async_cp16(bdstb + 48, bsrc + 48);
  wait_async0();
  __syncthreads();

  bf16x16 af0 = *(const bf16x16*)(asrc);        // A frag, stage 0 slice 0
  bf16x16 af1 = *(const bf16x16*)(asrc + 16);   // A frag, stage 0 slice 1

  const int nstage = K_TOT / BK2;               // 64
#pragma clang loop unroll(disable)
  for (int ki = 0; ki < nstage - 1; ++ki) {     // branch-free body; last peeled
    const int cur = ki & 1;

    // Async B staging for the next stage into the back buffer.
    const char* ns = bsrc + (size_t)(ki + 1) * (BK2 * 2);
    const unsigned d = bdstb + (unsigned)((cur ^ 1) * (TILE2_U * 4));
    async_cp16(d,      ns);
    async_cp16(d + 16, ns + 16);
    async_cp16(d + 32, ns + 32);
    async_cp16(d + 48, ns + 48);
    // Prefetch next A fragments.
    const unsigned* an = asrc + (size_t)(ki + 1) * (BK2 / 2);
    bf16x16 af0_n = *(const bf16x16*)(an);
    bf16x16 af1_n = *(const bf16x16*)(an + 16);

    const unsigned* lb = lds + cur * TILE2_U + bf_lane;
    wmma_step<LDS_STRIDE2 / 2>(af0, lb,      acc);   // K slice 0
    wmma_step<LDS_STRIDE2 / 2>(af1, lb + 16, acc);   // K slice 1
    af0 = af0_n;
    af1 = af1_n;

    wait_async0();        // own async writes complete before the barrier
    __syncthreads();
  }
  // Peeled final stage.
  {
    const unsigned* lb = lds + ((nstage - 1) & 1) * TILE2_U + bf_lane;
    wmma_step<LDS_STRIDE2 / 2>(af0, lb,      acc);
    wmma_step<LDS_STRIDE2 / 2>(af1, lb + 16, acc);
  }

  lstm_epilogue(acc, c0, bias, out_c, out_h, m0, n0, wv, l16, half);
}

// ==================================================================
// Fallback (workspace too small): fused f32->bf16 conversion in-kernel.
// ==================================================================
static __device__ __forceinline__ void store_tile_fb(unsigned* dst,
    const float4 r[4], int r0, int col)
{
#pragma unroll
  for (int p = 0; p < 4; ++p) {
    const int row = r0 + p * 32;
    uint2 v;
    v.x = pack2bf(r[p].x, r[p].y);
    v.y = pack2bf(r[p].z, r[p].w);
    *(uint2*)(dst + ((row * LDS_STRIDE + col) >> 1)) = v;
  }
}

__global__ __launch_bounds__(256)
void lstm_wmma_inline(const float* __restrict__ y,  const float* __restrict__ ctx,
                      const float* __restrict__ c0, const float* __restrict__ h0,
                      const float* __restrict__ W,  const float* __restrict__ U,
                      const float* __restrict__ Cw, const float* __restrict__ bias,
                      float* __restrict__ out_c,    float* __restrict__ out_h)
{
  __shared__ __attribute__((aligned(32))) unsigned lds[2 * BUF_U];   // 48 KB

  const int tid  = threadIdx.x;
  const int lane = tid & 31;
  const int wv   = tid >> 5;
  const int l16  = lane & 15;
  const int half = lane >> 4;
  const int m0   = blockIdx.y * BM;
  const int n0   = blockIdx.x * BN;
  const int r0   = tid >> 3;        // 0..31
  const int col  = (tid & 7) * 4;   // 0..28

  const unsigned aoff1 = (unsigned)(m0 + r0) * 1024u + col;   // y / h0
  const unsigned aoff2 = (unsigned)(m0 + r0) * 2048u + col;   // ctx
  const unsigned boff1 = (unsigned)(n0 + r0) * 1024u + col;   // W / U
  const unsigned boff2 = (unsigned)(n0 + r0) * 2048u + col;   // C

  f32x8 acc[8];
#pragma unroll
  for (int t = 0; t < 8; ++t)
    acc[t] = (f32x8){0.f,0.f,0.f,0.f,0.f,0.f,0.f,0.f};

  float4 ra[4], rb[4];
  auto stage_load = [&](int k0) {     // k0 is uniform -> scalar branches
    if (k0 < IN_DIM) {
      const float* a = y + aoff1 + k0;
      const float* b = W + boff1 + k0;
#pragma unroll
      for (int p = 0; p < 4; ++p) {
        ra[p] = *(const float4*)(a + (size_t)p * (32 * 1024));
        rb[p] = *(const float4*)(b + (size_t)p * (1024 * 1024));
      }
    } else if (k0 < IN_DIM + H_DIM) {
      const float* a = h0 + aoff1 + (k0 - IN_DIM);
      const float* b = U  + boff1 + (k0 - IN_DIM);
#pragma unroll
      for (int p = 0; p < 4; ++p) {
        ra[p] = *(const float4*)(a + (size_t)p * (32 * 1024));
        rb[p] = *(const float4*)(b + (size_t)p * (1024 * 1024));
      }
    } else {
      const float* a = ctx + aoff2 + (k0 - IN_DIM - H_DIM);
      const float* b = Cw  + boff2 + (k0 - IN_DIM - H_DIM);
#pragma unroll
      for (int p = 0; p < 4; ++p) {
        ra[p] = *(const float4*)(a + (size_t)p * (32 * 2048));
        rb[p] = *(const float4*)(b + (size_t)p * (1024 * 2048));
      }
    }
  };

  stage_load(0);
  store_tile_fb(&lds[0],      ra, r0, col);
  store_tile_fb(&lds[TILE_U], rb, r0, col);
  __syncthreads();

  const int a_fr    = (wv * 16 + l16) * (LDS_STRIDE / 2) + half * 8;
  const int bf_lane = l16 * (LDS_STRIDE / 2) + half * 8;

  const int nsteps = K_TOT / BK;   // 128
#pragma clang loop unroll(disable)
  for (int ki = 0; ki < nsteps - 1; ++ki) {
    const int cur = ki & 1;
    stage_load((ki + 1) * BK);

    const unsigned* la = &lds[cur * BUF_U];
    bf16x16 af = *(const bf16x16*)(la + a_fr);
    wmma_step<LDS_STRIDE / 2>(af, la + TILE_U + bf_lane, acc);

    unsigned* da = &lds[(cur ^ 1) * BUF_U];
    store_tile_fb(da,          ra, r0, col);
    store_tile_fb(da + TILE_U, rb, r0, col);
    __syncthreads();
  }
  {
    const unsigned* la = &lds[((nsteps - 1) & 1) * BUF_U];
    bf16x16 af = *(const bf16x16*)(la + a_fr);
    wmma_step<LDS_STRIDE / 2>(af, la + TILE_U + bf_lane, acc);
  }

  lstm_epilogue(acc, c0, bias, out_c, out_h, m0, n0, wv, l16, half);
}

// ==================================================================
extern "C" void kernel_launch(void* const* d_in, const int* in_sizes, int n_in,
                              void* d_out, int out_size, void* d_ws, size_t ws_size,
                              hipStream_t stream) {
  (void)in_sizes; (void)n_in; (void)out_size;
  const float* y   = (const float*)d_in[0];
  const float* ctx = (const float*)d_in[1];
  const float* c0  = (const float*)d_in[2];
  const float* h0  = (const float*)d_in[3];
  const float* W   = (const float*)d_in[4];
  const float* U   = (const float*)d_in[5];
  const float* Cw  = (const float*)d_in[6];
  const float* b   = (const float*)d_in[7];
  float* out_c = (float*)d_out;                       // tuple (c, h) concatenated
  float* out_h = out_c + (size_t)B_DIM * H_DIM;

  const size_t needA = (size_t)B_DIM * K_TOT * 2;         // 134 MB bf16
  const size_t needB = (size_t)(4 * H_DIM) * K_TOT * 2;   //  33 MB bf16
  const dim3 grid(H_DIM / BN, B_DIM / BM);                // (32, 128)

  if (ws_size >= needA + needB) {
    unsigned* Abf = (unsigned*)d_ws;
    unsigned* Bbf = Abf + needA / 4;
    convert_cat_bf16<<<(unsigned)((size_t)B_DIM * K_TOT / 4 / 256), 256, 0, stream>>>(
        y, h0, ctx, Abf);
    convert_cat_bf16<<<(unsigned)((size_t)(4 * H_DIM) * K_TOT / 4 / 256), 256, 0, stream>>>(
        W, U, Cw, Bbf);
    lstm_wmma_pre<<<grid, 256, 0, stream>>>(Abf, Bbf, c0, b, out_c, out_h);
  } else {
    lstm_wmma_inline<<<grid, 256, 0, stream>>>(y, ctx, c0, h0, W, U, Cw, b,
                                               out_c, out_h);
  }
}
